// BertEncoder_55972013801603
// MI455X (gfx1250) — matrix operand order repaired and checked by reading, not compiled, not run on hardware
//
#include <hip/hip_runtime.h>
#include <math.h>

typedef __bf16 bf16_t;
typedef __attribute__((ext_vector_type(16))) __bf16 bf16x16;
typedef __attribute__((ext_vector_type(8)))  __bf16 bf16x8;
typedef __attribute__((ext_vector_type(4)))  __bf16 bf16x4;
typedef __attribute__((ext_vector_type(8)))  float  f32x8;
typedef __attribute__((ext_vector_type(4)))  float  f32x4;

#define DEV static __device__ __forceinline__

// fp32 -> bf16 round-to-nearest-even
DEV bf16_t f2bf(float f) {
    unsigned u = __builtin_bit_cast(unsigned, f);
    unsigned r = u + 0x7FFFu + ((u >> 16) & 1u);
    unsigned short h = (unsigned short)(r >> 16);
    return __builtin_bit_cast(bf16_t, h);
}

DEV bf16x16 cat16(bf16x8 lo, bf16x8 hi) {
    return __builtin_shufflevector(lo, hi, 0, 1, 2, 3, 4, 5, 6, 7,
                                   8, 9, 10, 11, 12, 13, 14, 15);
}

// Low 32 bits of a generic pointer into LDS == workgroup-relative LDS byte
// address (ISA: generic-LDS addr = {SHARED_BASE[63:32], offset[31:0]}).
DEV unsigned lds_off_of(const void* p) {
    return (unsigned)(unsigned long long)p;
}

// GLOBAL_LOAD_ASYNC_TO_LDS_B128: per-lane 16-byte copy global -> LDS,
// tracked by ASYNCcnt (CDNA5 ISA).
DEV void async_b128(unsigned lds, const void* g) {
    asm volatile("global_load_async_to_lds_b128 %0, %1, off"
                 :: "v"(lds), "v"(g) : "memory");
}
DEV void wait_async0() {
    asm volatile("s_wait_asynccnt 0x0" ::: "memory");
}

// A fragment: 16(M) x 32(K) bf16 from row-major LDS tile ('stride' elems/row).
// Lanes 0-15 -> row=lane, K {0..7,16..23}; lanes 16-31 -> K {8..15,24..31}.
DEV bf16x16 load_a_frag(const bf16_t* base, int stride, int lane) {
    int half = (lane >> 4) & 1;
    int row  = lane & 15;
    const bf16_t* p = base + row * stride + half * 8;
    bf16x8 lo = *(const bf16x8*)(p);
    bf16x8 hi = *(const bf16x8*)(p + 16);
    return cat16(lo, hi);
}

// B fragment: 32(K) x 16(N) bf16 from column-major LDS (col n at base+n*cstride).
DEV bf16x16 load_b_frag(const bf16_t* base, int cstride, int lane) {
    int n  = lane & 15;
    int kh = (lane >> 4) & 1;
    const bf16_t* p = base + n * cstride + kh * 8;
    bf16x8 lo = *(const bf16x8*)(p);
    bf16x8 hi = *(const bf16x8*)(p + 16);
    return cat16(lo, hi);
}

DEV f32x8 wmma_bf16(bf16x16 a, bf16x16 b, f32x8 c) {
    return __builtin_amdgcn_wmma_f32_16x16x32_bf16(false, a, false, b,
                                                   (short)0, c, false, false);
}

// ---------------------------------------------------------------------------
// Weight prep: fp32 [L][K][N] -> bf16 transposed [L][N][K]
// ---------------------------------------------------------------------------
__global__ __launch_bounds__(256) void transpose_convert_kernel(
    const float* __restrict__ in, bf16_t* __restrict__ out, int K, int N)
{
    __shared__ float tile[32][33];
    const int kb = blockIdx.x * 32;
    const int nb = blockIdx.y * 32;
    const size_t lbase = (size_t)blockIdx.z * K * N;
    const int tx = threadIdx.x & 31;
    const int ty = threadIdx.x >> 5;
#pragma unroll
    for (int i = 0; i < 4; ++i) {
        int k = ty + i * 8;
        tile[k][tx] = in[lbase + (size_t)(kb + k) * N + nb + tx];
    }
    __syncthreads();
#pragma unroll
    for (int i = 0; i < 4; ++i) {
        int n = ty + i * 8;
        out[lbase + (size_t)(nb + n) * K + kb + tx] = f2bf(tile[tx][n]);
    }
}

// fp32 -> bf16 flat copy (initial hidden states)
__global__ __launch_bounds__(256) void convert_kernel(
    const float* __restrict__ in, bf16_t* __restrict__ out, int n4)
{
    int i = blockIdx.x * 256 + threadIdx.x;
    const int stride = gridDim.x * 256;
    for (; i < n4; i += stride) {
        f32x4 v = *(const f32x4*)(in + (size_t)i * 4);
        bf16x4 o;
        o[0] = f2bf(v.x); o[1] = f2bf(v.y); o[2] = f2bf(v.z); o[3] = f2bf(v.w);
        *(bf16x4*)(out + (size_t)i * 4) = o;
    }
}

// ---------------------------------------------------------------------------
// GEMM: A(bf16,[M][K]) @ W^T(bf16,[N][K]) + bias [+resid] [gelu]
//       -> Cf (fp32, opt) and/or Cb (bf16, opt)
// Block 128(M) x 128(N); 8 waves in 4x2 grid; wave tile 32x64; K-step 32.
// Double-buffered LDS tiles filled by GLOBAL_LOAD_ASYNC_TO_LDS_B128 with
// copy/compute overlap: one barrier + one asynccnt wait per K-step.
// ---------------------------------------------------------------------------
__global__ __launch_bounds__(256) void gemm_bias_kernel(
    const bf16_t* __restrict__ A, const bf16_t* __restrict__ Wt,
    const float* __restrict__ bias, const float* __restrict__ resid,
    float* __restrict__ Cf, bf16_t* __restrict__ Cb,
    int M, int N, int K, int do_gelu)
{
    __shared__ bf16_t As[2][128 * 32];   // [m][k]
    __shared__ bf16_t Bs[2][128 * 32];   // [n][k]

    const int tid  = threadIdx.x;
    const int lane = tid & 31;
    const int wave = tid >> 5;
    const int wm = wave >> 1;            // 0..3
    const int wn = wave & 1;             // 0..1
    const int m0 = blockIdx.x * 128;
    const int n0 = blockIdx.y * 128;

    f32x8 acc[2][4] = {};

    // Tile staging: 128x32 bf16 = 8KB per tile; 2 b128 asyncs per thread each.
    const int r = tid >> 1;              // 0..127
    const int c = (tid & 1) * 16;        // 0/16
    const bf16_t* agp = A  + (size_t)(m0 + r) * K + c;
    const bf16_t* bgp = Wt + (size_t)(n0 + r) * K + c;
    unsigned aoff[2], boff[2];
#pragma unroll
    for (int bfi = 0; bfi < 2; ++bfi) {
        aoff[bfi] = lds_off_of(&As[bfi][r * 32 + c]);
        boff[bfi] = lds_off_of(&Bs[bfi][r * 32 + c]);
    }

    auto issue = [&](int buf, int k0) {
        async_b128(aoff[buf],      agp + k0);
        async_b128(aoff[buf] + 16, agp + k0 + 8);
        async_b128(boff[buf],      bgp + k0);
        async_b128(boff[buf] + 16, bgp + k0 + 8);
    };

    const int T = K / 32;
    issue(0, 0);
    wait_async0();
    __syncthreads();

    for (int t = 0; t < T; ++t) {
        const int cur = t & 1;
        if (t + 1 < T) issue(cur ^ 1, (t + 1) * 32);
        if (t + 2 < T) {   // L2 prefetch two tiles ahead
            __builtin_prefetch(agp + (t + 2) * 32, 0, 1);
            __builtin_prefetch(bgp + (t + 2) * 32, 0, 1);
        }
        bf16x16 af[2], bfr[4];
#pragma unroll
        for (int mi = 0; mi < 2; ++mi)
            af[mi] = load_a_frag(&As[cur][(wm * 32 + mi * 16) * 32], 32, lane);
#pragma unroll
        for (int ni = 0; ni < 4; ++ni)
            bfr[ni] = load_b_frag(&Bs[cur][(wn * 64 + ni * 16) * 32], 32, lane);
#pragma unroll
        for (int mi = 0; mi < 2; ++mi)
#pragma unroll
            for (int ni = 0; ni < 4; ++ni)
                acc[mi][ni] = wmma_bf16(af[mi], bfr[ni], acc[mi][ni]);
        if (t + 1 < T) { wait_async0(); __syncthreads(); }
    }

    // C layout: VGPR i of lane -> row (lane<16 ? i : 8+i), col lane&15.
    const int colb = lane & 15;
    const int rsel = (lane >> 4) * 8;
#pragma unroll
    for (int mi = 0; mi < 2; ++mi) {
#pragma unroll
        for (int ni = 0; ni < 4; ++ni) {
            const int ncol = n0 + wn * 64 + ni * 16 + colb;
            const float bv = bias[ncol];
#pragma unroll
            for (int i = 0; i < 8; ++i) {
                const int mrow = m0 + wm * 32 + mi * 16 + rsel + i;
                float v = acc[mi][ni][i] + bv;
                if (resid) v += resid[(size_t)mrow * N + ncol];
                if (do_gelu) v = 0.5f * v * (1.0f + erff(v * 0.70710678118f));
                if (Cf) Cf[(size_t)mrow * N + ncol] = v;
                if (Cb) Cb[(size_t)mrow * N + ncol] = f2bf(v);
            }
        }
    }
}

// ---------------------------------------------------------------------------
// Flash attention, all-bf16 I/O. One block per (b, head, 128-row q tile);
// each wave owns 16 q rows. KV blocks of 32, online softmax, scale on logits.
// Q + K tiles via async LDS copies; K/V double-buffered for overlap.
// ---------------------------------------------------------------------------
__global__ __launch_bounds__(256) void attention_kernel(
    const bf16_t* __restrict__ Q, const bf16_t* __restrict__ Kg,
    const bf16_t* __restrict__ V, const float* __restrict__ mask,
    bf16_t* __restrict__ O)
{
    __shared__ bf16_t Qs[128 * 64];       // [q][d]
    __shared__ bf16_t Ks[2][32 * 64];     // [kv][d]
    __shared__ bf16_t Vt[2][64 * 32];     // [d][kv]
    __shared__ bf16_t Ps[8][16 * 32];     // per-wave P tile [q][kv]

    const int tid  = threadIdx.x;
    const int lane = tid & 31;
    const int wv   = tid >> 5;
    const int bid  = blockIdx.x;
    const int qt   = bid & 3;
    const int head = (bid >> 2) % 12;
    const int b    = bid / 48;
    const int q0   = qt * 128;
    const size_t baserow = (size_t)b * 512;
    const int hcol = head * 64;

    // Stage Q: 128x64 bf16 = 1024 x b128 chunks, 4 per thread, async.
#pragma unroll
    for (int it = 0; it < 4; ++it) {
        int idx = tid + it * 256;
        int rr = idx >> 3;
        int cc = (idx & 7) * 8;
        async_b128(lds_off_of(Qs + rr * 64 + cc),
                   Q + (baserow + q0 + rr) * 768 + hcol + cc);
    }

    const int kr = tid >> 3;        // 0..31
    const int kc = (tid & 7) * 8;   // 0..56

    auto stage_kv = [&](int buf, int j) {
        // K tile: one async b128 per thread.
        async_b128(lds_off_of(&Ks[buf][kr * 64 + kc]),
                   Kg + (baserow + j + kr) * 768 + hcol + kc);
        // V tile: load 8 bf16, transpose-scatter into Vt[d][kv].
        bf16x8 vv = *(const bf16x8*)(V + (baserow + j + kr) * 768 + hcol + kc);
#pragma unroll
        for (int i = 0; i < 8; ++i) Vt[buf][(kc + i) * 32 + kr] = vv[i];
    };

    f32x8 oacc[4] = {};
    float mrow[8], lrow[8];
#pragma unroll
    for (int i = 0; i < 8; ++i) { mrow[i] = -1e30f; lrow[i] = 0.0f; }

    stage_kv(0, 0);
    wait_async0();                  // Q + first K tile
    __syncthreads();                // + first V scatter visible

    for (int t = 0; t < 16; ++t) {
        const int cur = t & 1;
        const int j = t * 32;
        if (t + 1 < 16) stage_kv(cur ^ 1, j + 32);

        // S tile [16 q][32 kv]: two 16x16 WMMA tiles, contraction over d=64.
        f32x8 c0 = {}, c1 = {};
#pragma unroll
        for (int s = 0; s < 2; ++s) {
            bf16x16 a   = load_a_frag(Qs + (wv * 16) * 64 + s * 32, 64, lane);
            bf16x16 bk0 = load_b_frag(&Ks[cur][0 * 64 + s * 32], 64, lane);
            bf16x16 bk1 = load_b_frag(&Ks[cur][16 * 64 + s * 32], 64, lane);
            c0 = wmma_bf16(a, bk0, c0);
            c1 = wmma_bf16(a, bk1, c1);
        }
        const float mv0 = mask[b * 512 + j + (lane & 15)];
        const float mv1 = mask[b * 512 + j + 16 + (lane & 15)];
#pragma unroll
        for (int i = 0; i < 8; ++i) {
            float s0 = c0[i] * 0.125f + mv0;    // 1/sqrt(DH) on logits
            float s1 = c1[i] * 0.125f + mv1;
            float tmx = fmaxf(s0, s1);
#pragma unroll
            for (int o2 = 8; o2 >= 1; o2 >>= 1) tmx = fmaxf(tmx, __shfl_xor(tmx, o2, 32));
            float mn = fmaxf(mrow[i], tmx);
            float p0 = __expf(s0 - mn);
            float p1 = __expf(s1 - mn);
            float su = p0 + p1;
#pragma unroll
            for (int o2 = 8; o2 >= 1; o2 >>= 1) su += __shfl_xor(su, o2, 32);
            float alpha = __expf(mrow[i] - mn);
            lrow[i] = lrow[i] * alpha + su;
            mrow[i] = mn;
#pragma unroll
            for (int nt = 0; nt < 4; ++nt) oacc[nt][i] *= alpha;
            int rr = (lane < 16) ? i : 8 + i;
            Ps[wv][rr * 32 + (lane & 15)]      = f2bf(p0);
            Ps[wv][rr * 32 + 16 + (lane & 15)] = f2bf(p1);
        }
        // O += P @ V  (wave-private Ps; DS dependency via dscnt waits)
        bf16x16 ap = load_a_frag(Ps[wv], 32, lane);
#pragma unroll
        for (int nt = 0; nt < 4; ++nt) {
            bf16x16 bvf = load_b_frag(&Vt[cur][nt * 16 * 32], 32, lane);
            oacc[nt] = wmma_bf16(ap, bvf, oacc[nt]);
        }
        if (t + 1 < 16) { wait_async0(); __syncthreads(); }
    }
#pragma unroll
    for (int nt = 0; nt < 4; ++nt) {
#pragma unroll
        for (int i = 0; i < 8; ++i) {
            int rr = q0 + wv * 16 + ((lane < 16) ? i : 8 + i);
            int cidx = hcol + nt * 16 + (lane & 15);
            O[(baserow + rr) * 768 + cidx] = f2bf(oacc[nt][i] / lrow[i]);
        }
    }
}

// ---------------------------------------------------------------------------
// LayerNorm over H=768, eps=1e-12. Writes fp32 (residual path) + bf16 copy.
// ---------------------------------------------------------------------------
__global__ __launch_bounds__(256) void layernorm_kernel(
    const float* __restrict__ X, const float* __restrict__ gam,
    const float* __restrict__ bet, float* __restrict__ Yf,
    bf16_t* __restrict__ Yb)
{
    __shared__ float red[8];
    const int row = blockIdx.x;
    const int tid = threadIdx.x;
    const int lane = tid & 31;
    const int wv = tid >> 5;
    const float* x = X + (size_t)row * 768;

    float v0 = x[tid], v1 = x[tid + 256], v2 = x[tid + 512];
    float s = v0 + v1 + v2;
#pragma unroll
    for (int off = 16; off >= 1; off >>= 1) s += __shfl_xor(s, off, 32);
    if (lane == 0) red[wv] = s;
    __syncthreads();
    float tot = (lane < 8) ? red[lane] : 0.0f;
#pragma unroll
    for (int off = 4; off >= 1; off >>= 1) tot += __shfl_xor(tot, off, 32);
    tot = __shfl(tot, 0, 32);
    const float mu = tot * (1.0f / 768.0f);
    __syncthreads();

    float d0 = v0 - mu, d1 = v1 - mu, d2 = v2 - mu;
    float q = d0 * d0 + d1 * d1 + d2 * d2;
#pragma unroll
    for (int off = 16; off >= 1; off >>= 1) q += __shfl_xor(q, off, 32);
    if (lane == 0) red[wv] = q;
    __syncthreads();
    float qt = (lane < 8) ? red[lane] : 0.0f;
#pragma unroll
    for (int off = 4; off >= 1; off >>= 1) qt += __shfl_xor(qt, off, 32);
    qt = __shfl(qt, 0, 32);
    const float inv = rsqrtf(qt * (1.0f / 768.0f) + 1e-12f);

    size_t o = (size_t)row * 768;
    float y0 = d0 * inv * gam[tid]       + bet[tid];
    float y1 = d1 * inv * gam[tid + 256] + bet[tid + 256];
    float y2 = d2 * inv * gam[tid + 512] + bet[tid + 512];
    Yf[o + tid] = y0;       Yb[o + tid] = f2bf(y0);
    Yf[o + tid + 256] = y1; Yb[o + tid + 256] = f2bf(y1);
    Yf[o + tid + 512] = y2; Yb[o + tid + 512] = f2bf(y2);
}

// ---------------------------------------------------------------------------
// Host orchestration
// ---------------------------------------------------------------------------
extern "C" void kernel_launch(void* const* d_in, const int* in_sizes, int n_in,
                              void* d_out, int out_size, void* d_ws, size_t ws_size,
                              hipStream_t stream)
{
    (void)in_sizes; (void)n_in; (void)out_size; (void)ws_size;
    const int Lc = 12, Hc = 768, FFc = 3072;
    const int Mc = 8 * 512;  // B*S = 4096

    const float* hidden = (const float*)d_in[0];
    const float* amask  = (const float*)d_in[1];
    const float* Wq  = (const float*)d_in[2];
    const float* bq  = (const float*)d_in[3];
    const float* Wk  = (const float*)d_in[4];
    const float* bk  = (const float*)d_in[5];
    const float* Wv  = (const float*)d_in[6];
    const float* bv  = (const float*)d_in[7];
    const float* Wo  = (const float*)d_in[8];
    const float* bo  = (const float*)d_in[9];
    const float* g1  = (const float*)d_in[10];
    const float* b1  = (const float*)d_in[11];
    const float* Wi  = (const float*)d_in[12];
    const float* bi  = (const float*)d_in[13];
    const float* Wo2 = (const float*)d_in[14];
    const float* bo2 = (const float*)d_in[15];
    const float* g2  = (const float*)d_in[16];
    const float* b2  = (const float*)d_in[17];

    char* w = (char*)d_ws;
    size_t off = 0;
    auto alloc = [&](size_t bytes) -> void* {
        off = (off + 255) & ~(size_t)255;
        void* p = w + off;
        off += bytes;
        return p;
    };
    const size_t wqE = (size_t)Lc * Hc * Hc;
    const size_t wiE = (size_t)Lc * Hc * FFc;
    bf16_t* wqT  = (bf16_t*)alloc(wqE * 2);
    bf16_t* wkT  = (bf16_t*)alloc(wqE * 2);
    bf16_t* wvT  = (bf16_t*)alloc(wqE * 2);
    bf16_t* woT  = (bf16_t*)alloc(wqE * 2);
    bf16_t* wiT  = (bf16_t*)alloc(wiE * 2);
    bf16_t* wo2T = (bf16_t*)alloc(wiE * 2);
    const size_t szH  = (size_t)Mc * Hc * sizeof(float);
    const size_t szHb = (size_t)Mc * Hc * sizeof(bf16_t);
    float*  bufH   = (float*)alloc(szH);     // fp32 hidden (residual + output)
    float*  bufT   = (float*)alloc(szH);     // pre-LN sum
    float*  bufH1  = (float*)alloc(szH);     // LN1 out fp32 (residual)
    bf16_t* bufHb  = (bf16_t*)alloc(szHb);   // bf16 hidden (GEMM A)
    bf16_t* bufQb  = (bf16_t*)alloc(szHb);
    bf16_t* bufKb  = (bf16_t*)alloc(szHb);
    bf16_t* bufVb  = (bf16_t*)alloc(szHb);
    bf16_t* bufCb  = (bf16_t*)alloc(szHb);   // attention ctx
    bf16_t* bufH1b = (bf16_t*)alloc(szHb);   // LN1 out bf16
    bf16_t* bufIb  = (bf16_t*)alloc((size_t)Mc * FFc * sizeof(bf16_t));

    // One-time weight convert+transpose: [L][K][N] fp32 -> [L][N][K] bf16
    transpose_convert_kernel<<<dim3(24, 24, 12), 256, 0, stream>>>(Wq,  wqT,  Hc,  Hc);
    transpose_convert_kernel<<<dim3(24, 24, 12), 256, 0, stream>>>(Wk,  wkT,  Hc,  Hc);
    transpose_convert_kernel<<<dim3(24, 24, 12), 256, 0, stream>>>(Wv,  wvT,  Hc,  Hc);
    transpose_convert_kernel<<<dim3(24, 24, 12), 256, 0, stream>>>(Wo,  woT,  Hc,  Hc);
    transpose_convert_kernel<<<dim3(24, 96, 12), 256, 0, stream>>>(Wi,  wiT,  Hc,  FFc);
    transpose_convert_kernel<<<dim3(96, 24, 12), 256, 0, stream>>>(Wo2, wo2T, FFc, Hc);

    hipMemcpyAsync(bufH, hidden, szH, hipMemcpyDeviceToDevice, stream);
    convert_kernel<<<1024, 256, 0, stream>>>(hidden, bufHb, Mc * Hc / 4);

    const dim3 gH(Mc / 128, Hc / 128);    // (32, 6)
    const dim3 gF(Mc / 128, FFc / 128);   // (32, 24)

    for (int l = 0; l < Lc; ++l) {
        const bf16_t* wq_l  = wqT  + (size_t)l * Hc * Hc;
        const bf16_t* wk_l  = wkT  + (size_t)l * Hc * Hc;
        const bf16_t* wv_l  = wvT  + (size_t)l * Hc * Hc;
        const bf16_t* wo_l  = woT  + (size_t)l * Hc * Hc;
        const bf16_t* wi_l  = wiT  + (size_t)l * Hc * FFc;
        const bf16_t* wo2_l = wo2T + (size_t)l * Hc * FFc;

        gemm_bias_kernel<<<gH, 256, 0, stream>>>(bufHb, wq_l, bq + l * Hc, nullptr,
                                                 nullptr, bufQb, Mc, Hc, Hc, 0);
        gemm_bias_kernel<<<gH, 256, 0, stream>>>(bufHb, wk_l, bk + l * Hc, nullptr,
                                                 nullptr, bufKb, Mc, Hc, Hc, 0);
        gemm_bias_kernel<<<gH, 256, 0, stream>>>(bufHb, wv_l, bv + l * Hc, nullptr,
                                                 nullptr, bufVb, Mc, Hc, Hc, 0);
        attention_kernel<<<384, 256, 0, stream>>>(bufQb, bufKb, bufVb, amask, bufCb);
        gemm_bias_kernel<<<gH, 256, 0, stream>>>(bufCb, wo_l, bo + l * Hc, bufH,
                                                 bufT, nullptr, Mc, Hc, Hc, 0);
        layernorm_kernel<<<Mc, 256, 0, stream>>>(bufT, g1 + l * Hc, b1 + l * Hc,
                                                 bufH1, bufH1b);
        gemm_bias_kernel<<<gF, 256, 0, stream>>>(bufH1b, wi_l, bi + (size_t)l * FFc,
                                                 nullptr, nullptr, bufIb,
                                                 Mc, FFc, Hc, 1);
        gemm_bias_kernel<<<gH, 256, 0, stream>>>(bufIb, wo2_l, bo2 + l * Hc, bufH1,
                                                 bufT, nullptr, Mc, Hc, FFc, 0);
        layernorm_kernel<<<Mc, 256, 0, stream>>>(bufT, g2 + l * Hc, b2 + l * Hc,
                                                 bufH, bufHb);
    }

    hipMemcpyAsync(d_out, bufH, szH, hipMemcpyDeviceToDevice, stream);
}
